// Block_24747601560246
// MI455X (gfx1250) — matrix operand order repaired and checked by reading, not compiled, not run on hardware
//
#include <hip/hip_runtime.h>
#include <hip/hip_bf16.h>
#include <stdint.h>

#define B_  4
#define T_  2048
#define C_  768
#define H_  12
#define D_  64
#define BT_ (B_ * T_)
#define FF_ (4 * C_)
#define NQKV_ (3 * H_ * D_)   // 2304 fused QKV output columns

typedef __attribute__((ext_vector_type(16))) _Float16 v16h;
typedef __attribute__((ext_vector_type(8)))  float    v8f;
typedef __attribute__((ext_vector_type(8)))  uint32_t v8u;

// ---- WMMA fragment loaders (layouts per CDNA5 ISA 7.12.2, wave32) ----

// A-matrix 16x32 f16: lane l (m=l&15, g=l>>4); VGPR v holds
// K = 16*(v>>2) + 8*g + 2*(v&3) + {0,1}  (pairs contiguous in row-major A)
__device__ __forceinline__ v16h load_a_frag(const _Float16* __restrict__ A, int lda,
                                            int k0, int lane) {
  const int m = lane & 15, g = (lane >> 4) & 1;
  const uint32_t* p = reinterpret_cast<const uint32_t*>(A + (size_t)m * lda + k0);
  v8u w;
#pragma unroll
  for (int v = 0; v < 8; ++v) {
    const int kp = 16 * (v >> 2) + 8 * g + 2 * (v & 3);
    w[v] = p[kp >> 1];
  }
  return __builtin_bit_cast(v16h, w);
}

// B-matrix 32x16 f16: lane l (n=l&15, g=l>>4); VGPR v holds K = 16*(v>>2) + 8*g + 2*(v&3) + {0,1}.
// Bt stored transposed [N][K] so K-pairs are contiguous dwords.
__device__ __forceinline__ v16h load_b_frag(const _Float16* __restrict__ Bt, int ldb,
                                            int k0, int lane) {
  const int n = lane & 15, g = (lane >> 4) & 1;
  const uint32_t* p = reinterpret_cast<const uint32_t*>(Bt + (size_t)n * ldb + k0);
  v8u w;
#pragma unroll
  for (int v = 0; v < 8; ++v) w[v] = p[(16 * (v >> 2) + 8 * g + 2 * (v & 3)) >> 1];
  return __builtin_bit_cast(v16h, w);
}

__device__ __forceinline__ v8f wmma_f16(v16h a, v16h b, v8f c) {
  return __builtin_amdgcn_wmma_f32_16x16x32_f16(false, a, false, b, (short)0, c,
                                                false, false);
}

// Single-tile accumulate (used by attention): C += A[16xK] * Bt^T
__device__ __forceinline__ v8f gemm_acc(v8f c, const _Float16* __restrict__ A, int lda,
                                        const _Float16* __restrict__ Bt, int ldb,
                                        int K, int lane) {
  for (int k0 = 0; k0 < K; k0 += 32) {
    v16h a = load_a_frag(A, lda, k0, lane);
    v16h b = load_b_frag(Bt, ldb, k0, lane);
    c = wmma_f16(a, b, c);
  }
  return c;
}

// 2x2 register-blocked accumulate: 32x32 output per wave, 4 WMMAs per k-step
// (2 A-frag + 2 B-frag loads per 4 WMMAs -> half the VMEM per WMMA).
__device__ __forceinline__ void gemm_2x2(v8f c[4], const _Float16* __restrict__ A, int lda,
                                         const _Float16* __restrict__ Bt, int ldb,
                                         int K, int lane) {
  for (int k0 = 0; k0 < K; k0 += 32) {
    __builtin_prefetch(A + (size_t)(lane & 15) * lda + k0 + 256, 0, 1);
    v16h a0 = load_a_frag(A, lda, k0, lane);
    v16h a1 = load_a_frag(A + (size_t)16 * lda, lda, k0, lane);
    v16h b0 = load_b_frag(Bt, ldb, k0, lane);
    v16h b1 = load_b_frag(Bt + (size_t)16 * ldb, ldb, k0, lane);
    c[0] = wmma_f16(a0, b0, c[0]);
    c[1] = wmma_f16(a0, b1, c[1]);
    c[2] = wmma_f16(a1, b0, c[2]);
    c[3] = wmma_f16(a1, b1, c[3]);
  }
}

// Async-copy a 128-float bias segment into LDS (CDNA5 ASYNCcnt path, GVS mode:
// mem_addr = SADDR + VADDR32 + ioffset; VDST holds the LDS byte address).
__device__ __forceinline__ void stage_bias_async(const float* __restrict__ src,
                                                 float* lds, int colBase) {
  const int i = threadIdx.x & 127;
  const uint32_t ldsaddr = (uint32_t)(uintptr_t)(lds + i);  // low 32 bits = LDS offset
  const uint32_t voff = (uint32_t)((colBase + i) * 4);
  asm volatile("global_load_async_to_lds_b32 %0, %1, %2"
               :: "v"(ldsaddr), "v"(voff), "s"(src) : "memory");
  asm volatile("s_wait_asynccnt 0x0" ::: "memory");
  __syncthreads();
}

// ---- weight prep: fp32 -> f16, transposed to [N][K] ----

__global__ void __launch_bounds__(256) conv_qkv_w(const float* __restrict__ wq,
                                                  const float* __restrict__ wk,
                                                  const float* __restrict__ wv,
                                                  _Float16* __restrict__ wT) {
  int i = blockIdx.x * 256 + threadIdx.x;
  const int total = 3 * H_ * C_ * D_;
  if (i >= total) return;
  int mat = i / (H_ * C_ * D_); int r = i % (H_ * C_ * D_);
  int h = r / (C_ * D_); r %= (C_ * D_);
  int c = r / D_; int d = r % D_;
  const float* src = (mat == 0) ? wq : ((mat == 1) ? wk : wv);
  wT[(((size_t)mat * H_ + h) * D_ + d) * C_ + c] =
      (_Float16)src[((size_t)h * C_ + c) * D_ + d];
}

__global__ void __launch_bounds__(256) conv_wT(const float* __restrict__ w,
                                               _Float16* __restrict__ wT, int K, int N) {
  int i = blockIdx.x * 256 + threadIdx.x;
  if (i >= K * N) return;
  int k = i / N, n = i % N;
  wT[(size_t)n * K + k] = (_Float16)w[(size_t)k * N + n];
}

// ---- LayerNorm -> f16 ----

__global__ void __launch_bounds__(256) ln_kernel(const float* __restrict__ x,
                                                 const float* __restrict__ gg,
                                                 const float* __restrict__ bb,
                                                 _Float16* __restrict__ out) {
  __shared__ float s1[8], s2[8];
  __shared__ float smu, srstd;
  const int row = blockIdx.x;
  const int tid = threadIdx.x, lane = tid & 31, wid = tid >> 5;
  const float* xr = x + (size_t)row * C_;
  float s = 0.f, ss = 0.f;
  for (int i = tid; i < C_; i += 256) { float t = xr[i]; s += t; ss += t * t; }
#pragma unroll
  for (int m = 16; m >= 1; m >>= 1) {
    s += __shfl_xor(s, m, 32);
    ss += __shfl_xor(ss, m, 32);
  }
  if (lane == 0) { s1[wid] = s; s2[wid] = ss; }
  __syncthreads();
  if (tid == 0) {
    float a = 0.f, q = 0.f;
#pragma unroll
    for (int i = 0; i < 8; ++i) { a += s1[i]; q += s2[i]; }
    float mu = a / C_;
    float var = q / C_ - mu * mu;
    smu = mu;
    srstd = rsqrtf(var + 1e-5f);
  }
  __syncthreads();
  const float mu = smu, rstd = srstd;
  for (int i = tid; i < C_; i += 256)
    out[(size_t)row * C_ + i] = (_Float16)((xr[i] - mu) * rstd * gg[i] + bb[i]);
}

// ---- fused QKV GEMM: [BT,C] @ [C, 3*H*D]; block tile 64x128, wave 32x32 ----

__global__ void __launch_bounds__(256) qkv_gemm(const _Float16* __restrict__ h16,
                                                const _Float16* __restrict__ wT,
                                                _Float16* __restrict__ q16,
                                                _Float16* __restrict__ k16,
                                                _Float16* __restrict__ vT16) {
  const int lane = threadIdx.x & 31, wid = threadIdx.x >> 5;
  const int wm = wid >> 2, wn = wid & 3;  // 2x4 wave grid
  const int row0 = blockIdx.x * 64 + wm * 32;
  const int col0 = blockIdx.y * 128 + wn * 32;  // fused QKV column
  v8f c[4] = {{}, {}, {}, {}};
  gemm_2x2(c, h16 + (size_t)row0 * C_, C_, wT + (size_t)col0 * C_, C_, C_, lane);
  const int n = lane & 15, g = (lane >> 4) & 1;
#pragma unroll
  for (int cm = 0; cm < 2; ++cm)
#pragma unroll
    for (int cn = 0; cn < 2; ++cn) {
      const v8f& acc = c[cm * 2 + cn];
      const int tr0 = row0 + cm * 16;
      const int tc0 = col0 + cn * 16;               // within [0, 2304)
      const int mat = tc0 / (H_ * D_);
      const int hh = (tc0 % (H_ * D_)) / D_;
      const int d0 = tc0 % D_;
      if (mat == 2) {
#pragma unroll
        for (int r = 0; r < 8; ++r) {
          int row = tr0 + 8 * g + r;
          vT16[((size_t)hh * D_ + d0 + n) * BT_ + row] = (_Float16)acc[r];
        }
      } else {
        _Float16* dst = (mat == 0) ? q16 : k16;
#pragma unroll
        for (int r = 0; r < 8; ++r) {
          int row = tr0 + 8 * g + r;
          dst[((size_t)hh * BT_ + row) * D_ + d0 + n] = (_Float16)acc[r];
        }
      }
    }
}

// ---- flash-style causal attention: one wave per 16-query tile ----

__global__ void __launch_bounds__(256) attn_kernel(const _Float16* __restrict__ q16,
                                                   const _Float16* __restrict__ k16,
                                                   const _Float16* __restrict__ vT16,
                                                   _Float16* __restrict__ attn16) {
  __shared__ _Float16 plds[8][16 * 32];  // per-wave P staging (C-layout -> A-layout)
  const int lane = threadIdx.x & 31, wid = threadIdx.x >> 5;
  int tid = blockIdx.x * 8 + wid;
  const int tilesPerBH = T_ / 16;
  int b = tid / (H_ * tilesPerBH); int r0 = tid % (H_ * tilesPerBH);
  int h = r0 / tilesPerBH; int qt = r0 % tilesPerBH;
  const _Float16* Q  = q16 + ((size_t)h * BT_ + b * T_ + qt * 16) * D_;
  const _Float16* Kb = k16 + ((size_t)h * BT_ + b * T_) * D_;   // [s][d] == Bt layout
  const _Float16* Vb = vT16 + (size_t)h * D_ * BT_ + (size_t)b * T_;  // [d][t]
  const int n = lane & 15, g = (lane >> 4) & 1;
  float mrow[8], lrow[8];
  v8f o0 = {}, o1 = {}, o2 = {}, o3 = {};
#pragma unroll
  for (int r = 0; r < 8; ++r) { mrow[r] = -1e30f; lrow[r] = 0.f; }
  const int qend = qt * 16 + 16;  // causal: keys < qend
  for (int s0 = 0; s0 < qend; s0 += 32) {
    v8f sa = {};
    sa = gemm_acc(sa, Q, D_, Kb + (size_t)s0 * D_, D_, D_, lane);
    const bool haveB = (s0 + 16) < qend;
    v8f sb = {};
    if (haveB) sb = gemm_acc(sb, Q, D_, Kb + (size_t)(s0 + 16) * D_, D_, D_, lane);
    _Float16* pl = plds[wid];
#pragma unroll
    for (int r = 0; r < 8; ++r) {
      const int qrow = qt * 16 + 8 * g + r;
      float va = sa[r] * 0.125f;                 // HEAD_SIZE^-0.5
      if (s0 + n > qrow) va = -1e30f;            // causal mask
      float vb = haveB ? sb[r] * 0.125f : -1e30f;
      if (s0 + 16 + n > qrow) vb = -1e30f;
      float mx = fmaxf(va, vb);                  // row reduce in 16-lane half (wave32)
      mx = fmaxf(mx, __shfl_xor(mx, 1, 32));
      mx = fmaxf(mx, __shfl_xor(mx, 2, 32));
      mx = fmaxf(mx, __shfl_xor(mx, 4, 32));
      mx = fmaxf(mx, __shfl_xor(mx, 8, 32));
      const float mnew = fmaxf(mrow[r], mx);
      const float ea = __expf(va - mnew), eb = __expf(vb - mnew);
      float s = ea + eb;
      s += __shfl_xor(s, 1, 32);
      s += __shfl_xor(s, 2, 32);
      s += __shfl_xor(s, 4, 32);
      s += __shfl_xor(s, 8, 32);
      const float alpha = __expf(mrow[r] - mnew);
      lrow[r] = lrow[r] * alpha + s;
      mrow[r] = mnew;
      o0[r] *= alpha; o1[r] *= alpha; o2[r] *= alpha; o3[r] *= alpha;
      const int prow = 8 * g + r;
      pl[prow * 32 + n]      = (_Float16)ea;
      pl[prow * 32 + 16 + n] = (_Float16)eb;
    }
    asm volatile("s_wait_dscnt 0x0" ::: "memory");  // LDS writes visible wave-wide
    v16h pf  = load_a_frag(pl, 32, 0, lane);        // P as A-fragment (16x32)
    v16h vb0 = load_b_frag(Vb + (size_t)0 * 16 * BT_ + s0, BT_, 0, lane);
    v16h vb1 = load_b_frag(Vb + (size_t)1 * 16 * BT_ + s0, BT_, 0, lane);
    v16h vb2 = load_b_frag(Vb + (size_t)2 * 16 * BT_ + s0, BT_, 0, lane);
    v16h vb3 = load_b_frag(Vb + (size_t)3 * 16 * BT_ + s0, BT_, 0, lane);
    o0 = wmma_f16(pf, vb0, o0);
    o1 = wmma_f16(pf, vb1, o1);
    o2 = wmma_f16(pf, vb2, o2);
    o3 = wmma_f16(pf, vb3, o3);
    asm volatile("s_wait_dscnt 0x0" ::: "memory");  // done with LDS before rewrite
  }
#pragma unroll
  for (int r = 0; r < 8; ++r) {
    const size_t row = (size_t)b * T_ + qt * 16 + 8 * g + r;
    const float inv = 1.f / lrow[r];
    attn16[row * C_ + h * D_ + 0 * 16 + n] = (_Float16)(o0[r] * inv);
    attn16[row * C_ + h * D_ + 1 * 16 + n] = (_Float16)(o1[r] * inv);
    attn16[row * C_ + h * D_ + 2 * 16 + n] = (_Float16)(o2[r] * inv);
    attn16[row * C_ + h * D_ + 3 * 16 + n] = (_Float16)(o3[r] * inv);
  }
}

// ---- proj + residual: block 64x128, wave 32x32 ----

__global__ void __launch_bounds__(256) proj_kernel(const _Float16* __restrict__ attn16,
                                                   const _Float16* __restrict__ wpT,
                                                   const float* __restrict__ bproj,
                                                   const float* __restrict__ x,
                                                   float* __restrict__ x1) {
  __shared__ float bsh[128];
  const int lane = threadIdx.x & 31, wid = threadIdx.x >> 5;
  const int wm = wid >> 2, wn = wid & 3;
  const int row0 = blockIdx.x * 64 + wm * 32;
  const int col0 = blockIdx.y * 128 + wn * 32;
  stage_bias_async(bproj, bsh, blockIdx.y * 128);
  v8f c[4] = {{}, {}, {}, {}};
  gemm_2x2(c, attn16 + (size_t)row0 * C_, C_, wpT + (size_t)col0 * C_, C_, C_, lane);
  const int n = lane & 15, g = (lane >> 4) & 1;
#pragma unroll
  for (int cm = 0; cm < 2; ++cm)
#pragma unroll
    for (int cn = 0; cn < 2; ++cn) {
      const v8f& acc = c[cm * 2 + cn];
#pragma unroll
      for (int r = 0; r < 8; ++r) {
        int row = row0 + cm * 16 + 8 * g + r;
        int col = col0 + cn * 16 + n;
        x1[(size_t)row * C_ + col] =
            acc[r] + bsh[col - blockIdx.y * 128] + x[(size_t)row * C_ + col];
      }
    }
}

// ---- FFN1: relu(h2 @ w1 + b1) -> f16; block 64x128, wave 32x32 ----

__global__ void __launch_bounds__(256) ffn1_kernel(const _Float16* __restrict__ h2,
                                                   const _Float16* __restrict__ w1T,
                                                   const float* __restrict__ b1,
                                                   _Float16* __restrict__ mid) {
  __shared__ float bsh[128];
  const int lane = threadIdx.x & 31, wid = threadIdx.x >> 5;
  const int wm = wid >> 2, wn = wid & 3;
  const int row0 = blockIdx.x * 64 + wm * 32;
  const int col0 = blockIdx.y * 128 + wn * 32;
  stage_bias_async(b1, bsh, blockIdx.y * 128);
  v8f c[4] = {{}, {}, {}, {}};
  gemm_2x2(c, h2 + (size_t)row0 * C_, C_, w1T + (size_t)col0 * C_, C_, C_, lane);
  const int n = lane & 15, g = (lane >> 4) & 1;
#pragma unroll
  for (int cm = 0; cm < 2; ++cm)
#pragma unroll
    for (int cn = 0; cn < 2; ++cn) {
      const v8f& acc = c[cm * 2 + cn];
#pragma unroll
      for (int r = 0; r < 8; ++r) {
        int row = row0 + cm * 16 + 8 * g + r;
        int col = col0 + cn * 16 + n;
        mid[(size_t)row * FF_ + col] =
            (_Float16)fmaxf(acc[r] + bsh[col - blockIdx.y * 128], 0.f);
      }
    }
}

// ---- FFN2 + residual -> final output; block 64x128, wave 32x32, K=3072 ----

__global__ void __launch_bounds__(256) ffn2_kernel(const _Float16* __restrict__ mid,
                                                   const _Float16* __restrict__ w2T,
                                                   const float* __restrict__ b2,
                                                   const float* __restrict__ x1,
                                                   float* __restrict__ out) {
  __shared__ float bsh[128];
  const int lane = threadIdx.x & 31, wid = threadIdx.x >> 5;
  const int wm = wid >> 2, wn = wid & 3;
  const int row0 = blockIdx.x * 64 + wm * 32;
  const int col0 = blockIdx.y * 128 + wn * 32;
  stage_bias_async(b2, bsh, blockIdx.y * 128);
  v8f c[4] = {{}, {}, {}, {}};
  gemm_2x2(c, mid + (size_t)row0 * FF_, FF_, w2T + (size_t)col0 * FF_, FF_, FF_, lane);
  const int n = lane & 15, g = (lane >> 4) & 1;
#pragma unroll
  for (int cm = 0; cm < 2; ++cm)
#pragma unroll
    for (int cn = 0; cn < 2; ++cn) {
      const v8f& acc = c[cm * 2 + cn];
#pragma unroll
      for (int r = 0; r < 8; ++r) {
        int row = row0 + cm * 16 + 8 * g + r;
        int col = col0 + cn * 16 + n;
        out[(size_t)row * C_ + col] =
            x1[(size_t)row * C_ + col] + acc[r] + bsh[col - blockIdx.y * 128];
      }
    }
}

extern "C" void kernel_launch(void* const* d_in, const int* in_sizes, int n_in,
                              void* d_out, int out_size, void* d_ws, size_t ws_size,
                              hipStream_t stream) {
  const float* x      = (const float*)d_in[0];
  const float* wq     = (const float*)d_in[1];
  const float* wk     = (const float*)d_in[2];
  const float* wv     = (const float*)d_in[3];
  const float* w_proj = (const float*)d_in[4];
  const float* b_proj = (const float*)d_in[5];
  const float* w1     = (const float*)d_in[6];
  const float* b1     = (const float*)d_in[7];
  const float* w2     = (const float*)d_in[8];
  const float* b2     = (const float*)d_in[9];
  const float* ln1_g  = (const float*)d_in[10];
  const float* ln1_b  = (const float*)d_in[11];
  const float* ln2_g  = (const float*)d_in[12];
  const float* ln2_b  = (const float*)d_in[13];
  float* out = (float*)d_out;

  char* p = (char*)d_ws;
  auto alloc = [&](size_t bytes) -> char* {
    char* r = p;
    p += (bytes + 255) & ~size_t(255);
    return r;
  };
  _Float16* h16    = (_Float16*)alloc((size_t)BT_ * C_ * 2);
  _Float16* q16    = (_Float16*)alloc((size_t)H_ * BT_ * D_ * 2);
  _Float16* k16    = (_Float16*)alloc((size_t)H_ * BT_ * D_ * 2);
  _Float16* vT16   = (_Float16*)alloc((size_t)H_ * D_ * BT_ * 2);
  _Float16* attn16 = (_Float16*)alloc((size_t)BT_ * C_ * 2);
  float*    x1     = (float*)alloc((size_t)BT_ * C_ * 4);
  _Float16* h2_16  = (_Float16*)alloc((size_t)BT_ * C_ * 2);
  _Float16* mid16  = (_Float16*)alloc((size_t)BT_ * FF_ * 2);
  _Float16* wqkvT  = (_Float16*)alloc((size_t)3 * H_ * D_ * C_ * 2);
  _Float16* wpT    = (_Float16*)alloc((size_t)C_ * C_ * 2);
  _Float16* w1T    = (_Float16*)alloc((size_t)C_ * FF_ * 2);
  _Float16* w2T    = (_Float16*)alloc((size_t)FF_ * C_ * 2);

  // weight prep
  {
    int tot = 3 * H_ * C_ * D_;
    conv_qkv_w<<<(tot + 255) / 256, 256, 0, stream>>>(wq, wk, wv, wqkvT);
    conv_wT<<<(C_ * C_ + 255) / 256, 256, 0, stream>>>(w_proj, wpT, C_, C_);
    conv_wT<<<(C_ * FF_ + 255) / 256, 256, 0, stream>>>(w1, w1T, C_, FF_);
    conv_wT<<<(FF_ * C_ + 255) / 256, 256, 0, stream>>>(w2, w2T, FF_, C_);
  }
  // attention sublayer
  ln_kernel<<<BT_, 256, 0, stream>>>(x, ln1_g, ln1_b, h16);
  qkv_gemm<<<dim3(BT_ / 64, NQKV_ / 128), 256, 0, stream>>>(h16, wqkvT, q16, k16, vT16);
  attn_kernel<<<(B_ * H_ * (T_ / 16)) / 8, 256, 0, stream>>>(q16, k16, vT16, attn16);
  proj_kernel<<<dim3(BT_ / 64, C_ / 128), 256, 0, stream>>>(attn16, wpT, b_proj, x, x1);
  // FFN sublayer
  ln_kernel<<<BT_, 256, 0, stream>>>(x1, ln2_g, ln2_b, h2_16);
  ffn1_kernel<<<dim3(BT_ / 64, FF_ / 128), 256, 0, stream>>>(h2_16, w1T, b1, mid16);
  ffn2_kernel<<<dim3(BT_ / 64, C_ / 128), 256, 0, stream>>>(mid16, w2T, b2, x1, out);
}
